// MultiheadDilatedAttention_712964571627
// MI455X (gfx1250) — matrix operand order repaired and checked by reading, not compiled, not run on hardware
//
#include <hip/hip_runtime.h>
#include <cstdint>
#include <cstddef>

// ---------------------------------------------------------------------------
// MI455X / gfx1250: wave32, WMMA 16x16x32 f16 (f32 accum).
// MultiheadDilatedAttention (LongNet), B=2 N=8192 E=768 H=12 D=64.
// All matrix math runs through v_wmma_f32_16x16x32_f16.
// ---------------------------------------------------------------------------

typedef __attribute__((ext_vector_type(16))) _Float16 v16h;
typedef __attribute__((ext_vector_type(8)))  _Float16 v8h;
typedef __attribute__((ext_vector_type(8)))  float    v8f;
typedef __attribute__((ext_vector_type(4)))  float    v4f;

#define E_   768
#define M_   16384     // B*N
#define NSEQ 8192
#define NB   2
#define NH   12
#define HD   64

// ------------------------- helpers -----------------------------------------

// A-matrix fragment (16x32 f16). Per ISA: lane m = lane%16; with off = 8*(lane/16),
// vector elements 0..7 hold K = off..off+7, elements 8..15 hold K = off+16..off+23.
// 'p' points at row_base + k0 + off.
static __device__ inline v16h load_a_frag(const float* __restrict__ p) {
  v4f x0 = *(const v4f*)(p);
  v4f x1 = *(const v4f*)(p + 4);
  v4f x2 = *(const v4f*)(p + 16);
  v4f x3 = *(const v4f*)(p + 20);
  v16h a;
#pragma unroll
  for (int e = 0; e < 4; ++e) {
    a[e]      = (_Float16)x0[e];
    a[4 + e]  = (_Float16)x1[e];
    a[8 + e]  = (_Float16)x2[e];
    a[12 + e] = (_Float16)x3[e];
  }
  return a;
}

static __device__ inline v16h load_a_frag(const _Float16* __restrict__ p) {
  v8h c0 = *(const v8h*)(p);
  v8h c1 = *(const v8h*)(p + 16);
  v16h a;
#pragma unroll
  for (int e = 0; e < 8; ++e) { a[e] = c0[e]; a[8 + e] = c1[e]; }
  return a;
}

// ------------------------- f32 -> f16 convert -------------------------------

__global__ void cvt_f32_f16_kernel(const float* __restrict__ in,
                                   _Float16* __restrict__ out, int n) {
  int i = blockIdx.x * blockDim.x + threadIdx.x;
  int stride = gridDim.x * blockDim.x;
  for (; i < n; i += stride) out[i] = (_Float16)in[i];
}

__global__ void zero_u32_kernel(uint32_t* __restrict__ p, int n) {
  int i = blockIdx.x * blockDim.x + threadIdx.x;
  int stride = gridDim.x * blockDim.x;
  for (; i < n; i += stride) p[i] = 0u;
}

// ------------------------- GEMM: Y = X @ W^T + bias -------------------------
// X: [M_, E_] (TIn), W: [E_, E_] f16 row-major (used as B(k,n)=W[n,k]),
// Y: [M_, E_] (TOut). Block = 8 waves, wave tile 32x32 (2x2 WMMA accums).
template <typename TIn, typename TOut>
__global__ __launch_bounds__(256)
void gemm_wt_kernel(const TIn* __restrict__ X, const _Float16* __restrict__ W,
                    const float* __restrict__ bias, TOut* __restrict__ Y) {
  const int w    = threadIdx.x >> 5;
  const int lane = threadIdx.x & 31;
  const int hlf  = lane >> 4;
  const int lm   = lane & 15;
  const int wm = w & 3, wn = w >> 2;
  const int row0 = blockIdx.y * 128 + wm * 32;
  const int col0 = blockIdx.x * 64  + wn * 32;

  v8f acc[2][2] = {};
  const float bv0 = bias[col0 + lm];
  const float bv1 = bias[col0 + 16 + lm];
  const TIn* xr0 = X + (size_t)(row0 + lm) * E_;
  const TIn* xr1 = X + (size_t)(row0 + 16 + lm) * E_;
  const _Float16* wr0 = W + (size_t)(col0 + lm) * E_;
  const _Float16* wr1 = W + (size_t)(col0 + 16 + lm) * E_;

  for (int k0 = 0; k0 < E_; k0 += 32) {
    const int ab = k0 + hlf * 8;    // A chunk base
    v16h a0 = load_a_frag(xr0 + ab);
    v16h a1 = load_a_frag(xr1 + ab);
    const int bb = ab;              // B: same K chunks as A
    v16h b0 = load_a_frag(wr0 + bb);
    v16h b1 = load_a_frag(wr1 + bb);
    acc[0][0] = __builtin_amdgcn_wmma_f32_16x16x32_f16(false, a0, false, b0, (short)0, acc[0][0], false, false);
    acc[0][1] = __builtin_amdgcn_wmma_f32_16x16x32_f16(false, a0, false, b1, (short)0, acc[0][1], false, false);
    acc[1][0] = __builtin_amdgcn_wmma_f32_16x16x32_f16(false, a1, false, b0, (short)0, acc[1][0], false, false);
    acc[1][1] = __builtin_amdgcn_wmma_f32_16x16x32_f16(false, a1, false, b1, (short)0, acc[1][1], false, false);
  }

  // C/D layout: lane col = lane%16, VGPR v -> row v + 8*(lane/16)
#pragma unroll
  for (int mt = 0; mt < 2; ++mt)
#pragma unroll
    for (int nt = 0; nt < 2; ++nt) {
      const float bb = nt ? bv1 : bv0;
      const int col = col0 + nt * 16 + lm;
#pragma unroll
      for (int v = 0; v < 8; ++v) {
        const int row = row0 + mt * 16 + v + hlf * 8;
        Y[(size_t)row * E_ + col] = (TOut)(acc[mt][nt][v] + bb);
      }
    }
}

// ------------------------- Dilated flash attention --------------------------
// All groups have effective segment size 2048 (g/r). One block (4 waves) per
// (batch, group, segment, head, 64-query-block). Computes S^T = K x Q^T so the
// softmax reduction is per-lane and S^T's C-layout coincides with P's A-layout
// for the PxV WMMA. V chunk staged transposed in LDS (stride 40 halves).
__global__ __launch_bounds__(128)
void dilated_attn_kernel(const _Float16* __restrict__ Q,
                         const _Float16* __restrict__ K,
                         const _Float16* __restrict__ V,
                         _Float16* __restrict__ O) {
  int bid = blockIdx.x;
  const int qb = bid & 31; bid >>= 5;
  const int hg = bid & 3;  bid >>= 2;
  const int slot = bid % 7;
  const int b    = bid / 7;
  int grp, seg;
  if (slot < 4)      { grp = 0; seg = slot; }
  else if (slot < 6) { grp = 1; seg = slot - 4; }
  else               { grp = 2; seg = 0; }
  const int SEGL[3] = {2048, 4096, 8192};
  const int DIL[3]  = {1, 2, 4};
  const int g = SEGL[grp], r = DIL[grp], off = grp;   // offset = grp % r == grp
  const int head = grp * 4 + hg;
  const int segbase = seg * g;

  const int w    = threadIdx.x >> 5;
  const int lane = threadIdx.x & 31;
  const int hlf  = lane >> 4;
  const int lm   = lane & 15;

  const size_t baseBH = (size_t)b * NSEQ * E_ + (size_t)head * HD;

  // Q^T B-frags for this wave's 16 query rows (kept in regs for whole loop)
  const int mrow0 = qb * 64 + w * 16;
  const int tokq = segbase + off + (mrow0 + lm) * r;
  v16h qf[2];
#pragma unroll
  for (int ks = 0; ks < 2; ++ks)
    qf[ks] = load_a_frag(Q + baseBH + (size_t)tokq * E_ + ks * 32 + hlf * 8);

  v8f accO[4] = {};
  float run_max = -INFINITY, run_sum = 0.f;
  const float scale = 0.125f;  // 1/sqrt(64)

  __shared__ _Float16 VT[64 * 40];  // V^T chunk: [d=64][t=32], stride 40 halves

  for (int tc = 0; tc < 2048; tc += 32) {
    __syncthreads();
    {  // stage V^T: wave w handles d-range [w*16, w*16+16); lane = t-local
      const int tokv = segbase + off + (tc + lane) * r;
      v16h vv = *(const v16h*)(V + baseBH + (size_t)tokv * E_ + w * 16);
#pragma unroll
      for (int e = 0; e < 16; ++e) VT[(w * 16 + e) * 40 + lane] = vv[e];
    }
    __syncthreads();

    // S^T tiles: t-tile0 (t=tc..tc+15) and t-tile1 (t=tc+16..tc+31)
    v8f s0 = {}, s1 = {};
#pragma unroll
    for (int tt = 0; tt < 2; ++tt) {
      const int tokk = segbase + off + (tc + tt * 16 + lm) * r;
      const _Float16* kp = K + baseBH + (size_t)tokk * E_;
#pragma unroll
      for (int ks = 0; ks < 2; ++ks) {
        const _Float16* kpp = kp + ks * 32 + hlf * 8;
        v8h c0 = *(const v8h*)(kpp);
        v8h c1 = *(const v8h*)(kpp + 16);
        v16h ak;
#pragma unroll
        for (int e = 0; e < 8; ++e) { ak[e] = c0[e]; ak[8 + e] = c1[e]; }
        if (tt == 0)
          s0 = __builtin_amdgcn_wmma_f32_16x16x32_f16(false, ak, false, qf[ks], (short)0, s0, false, false);
        else
          s1 = __builtin_amdgcn_wmma_f32_16x16x32_f16(false, ak, false, qf[ks], (short)0, s1, false, false);
      }
    }
#pragma unroll
    for (int v = 0; v < 8; ++v) { s0[v] = s0[v] * scale; s1[v] = s1[v] * scale; }

    // online softmax; lane's column m = lane%16; lanes l and l^16 split the t's
    float cmax = s0[0];
#pragma unroll
    for (int v = 0; v < 8; ++v) { cmax = fmaxf(cmax, s0[v]); cmax = fmaxf(cmax, s1[v]); }
    cmax = fmaxf(cmax, __shfl_xor(cmax, 16, 32));
    const float nmax = fmaxf(run_max, cmax);
    const float corr = __expf(run_max - nmax);
    float p0[8], p1[8], csum = 0.f;
#pragma unroll
    for (int v = 0; v < 8; ++v) {
      p0[v] = __expf(s0[v] - nmax);
      p1[v] = __expf(s1[v] - nmax);
      csum += p0[v] + p1[v];
    }
    csum += __shfl_xor(csum, 16, 32);
    run_sum = run_sum * corr + csum;
    run_max = nmax;

    // P A-frag: S^T C-layout already matches A-layout element-for-element
    v16h pa;
#pragma unroll
    for (int v = 0; v < 8; ++v) { pa[v] = (_Float16)p0[v]; pa[8 + v] = (_Float16)p1[v]; }

    // rescale accO rows: row m = v + 8*hlf lives in lane (v + 8*hlf)
    float corrv[8];
#pragma unroll
    for (int v = 0; v < 8; ++v) corrv[v] = __shfl(corr, v + hlf * 8, 32);

#pragma unroll
    for (int dt = 0; dt < 4; ++dt) {
#pragma unroll
      for (int v = 0; v < 8; ++v) accO[dt][v] = accO[dt][v] * corrv[v];
      const int dcol = dt * 16 + lm;
      const _Float16* vp = &VT[dcol * 40 + hlf * 8];
      v8h b0 = *(const v8h*)(vp);
      v8h b1 = *(const v8h*)(vp + 16);
      v16h vb;
#pragma unroll
      for (int e = 0; e < 8; ++e) { vb[e] = b0[e]; vb[8 + e] = b1[e]; }
      accO[dt] = __builtin_amdgcn_wmma_f32_16x16x32_f16(false, pa, false, vb, (short)0, accO[dt], false, false);
    }
  }

  // epilogue: divide by row-sum, scale by 1/G, scatter to output
  const float linv = 1.f / run_sum;
  float lv[8];
#pragma unroll
  for (int v = 0; v < 8; ++v) lv[v] = __shfl(linv, v + hlf * 8, 32);
  const float ginv = (1.f / 3.f);
#pragma unroll
  for (int dt = 0; dt < 4; ++dt) {
    const int dcol = dt * 16 + lm;
#pragma unroll
    for (int v = 0; v < 8; ++v) {
      const int mabs = mrow0 + v + hlf * 8;
      const int tok  = segbase + off + mabs * r;
      O[baseBH + (size_t)tok * E_ + dcol] = (_Float16)(accO[dt][v] * lv[v] * ginv);
    }
  }
}

// ------------------------- LayerNorm ----------------------------------------
__global__ __launch_bounds__(256)
void ln_kernel(const _Float16* __restrict__ xin, const float* __restrict__ gam,
               const float* __restrict__ bet, _Float16* __restrict__ xout) {
  const int tok = blockIdx.x;
  const int tid = threadIdx.x;
  const _Float16* xr = xin + (size_t)tok * E_;
  const float v0 = (float)xr[tid];
  const float v1 = (float)xr[tid + 256];
  const float v2 = (float)xr[tid + 512];
  __shared__ float red[256];
  red[tid] = v0 + v1 + v2;
  __syncthreads();
  for (int o = 128; o > 0; o >>= 1) { if (tid < o) red[tid] += red[tid + o]; __syncthreads(); }
  const float mu = red[0] * (1.f / E_);
  __syncthreads();
  red[tid] = v0 * v0 + v1 * v1 + v2 * v2;
  __syncthreads();
  for (int o = 128; o > 0; o >>= 1) { if (tid < o) red[tid] += red[tid + o]; __syncthreads(); }
  const float var = red[0] * (1.f / E_) - mu * mu;
  const float rstd = rsqrtf(var + 1e-5f);
  _Float16* yr = xout + (size_t)tok * E_;
  yr[tid]       = (_Float16)((v0 - mu) * rstd * gam[tid]       + bet[tid]);
  yr[tid + 256] = (_Float16)((v1 - mu) * rstd * gam[tid + 256] + bet[tid + 256]);
  yr[tid + 512] = (_Float16)((v2 - mu) * rstd * gam[tid + 512] + bet[tid + 512]);
}

// ------------------------- launcher -----------------------------------------
extern "C" void kernel_launch(void* const* d_in, const int* in_sizes, int n_in,
                              void* d_out, int out_size, void* d_ws, size_t ws_size,
                              hipStream_t stream) {
  const float* query = (const float*)d_in[0];
  const float* key_  = (const float*)d_in[1];
  const float* value = (const float*)d_in[2];
  const float* Wq = (const float*)d_in[3];  const float* bq = (const float*)d_in[4];
  const float* Wk = (const float*)d_in[5];  const float* bk = (const float*)d_in[6];
  const float* Wv = (const float*)d_in[7];  const float* bv = (const float*)d_in[8];
  const float* Wo = (const float*)d_in[9];  const float* bo = (const float*)d_in[10];
  const float* ln_g = (const float*)d_in[11];
  const float* ln_b = (const float*)d_in[12];

  // workspace layout (~106 MB)
  char* ws = (char*)d_ws;
  const size_t szW = (size_t)E_ * E_ * sizeof(_Float16);   // 1,179,648
  const size_t szX = (size_t)M_ * E_ * sizeof(_Float16);   // 25,165,824
  _Float16* W16q = (_Float16*)(ws);
  _Float16* W16k = (_Float16*)(ws + szW);
  _Float16* W16v = (_Float16*)(ws + 2 * szW);
  _Float16* W16o = (_Float16*)(ws + 3 * szW);
  _Float16* Qf   = (_Float16*)(ws + 4 * szW);
  _Float16* Kf   = (_Float16*)(ws + 4 * szW + szX);
  _Float16* Vf   = (_Float16*)(ws + 4 * szW + 2 * szX);
  _Float16* Af   = (_Float16*)(ws + 4 * szW + 3 * szX);   // attention output (f16)

  const int nW = E_ * E_;
  cvt_f32_f16_kernel<<<1152, 256, 0, stream>>>(Wq, W16q, nW);
  cvt_f32_f16_kernel<<<1152, 256, 0, stream>>>(Wk, W16k, nW);
  cvt_f32_f16_kernel<<<1152, 256, 0, stream>>>(Wv, W16v, nW);
  cvt_f32_f16_kernel<<<1152, 256, 0, stream>>>(Wo, W16o, nW);

  dim3 ggrid(E_ / 64, M_ / 128);  // (12, 128)
  gemm_wt_kernel<float, _Float16><<<ggrid, 256, 0, stream>>>(query, W16q, bq, Qf);
  gemm_wt_kernel<float, _Float16><<<ggrid, 256, 0, stream>>>(key_,  W16k, bk, Kf);
  gemm_wt_kernel<float, _Float16><<<ggrid, 256, 0, stream>>>(value, W16v, bv, Vf);

  // zero attention buffer (dilated groups leave holes that must read as 0)
  zero_u32_kernel<<<4096, 256, 0, stream>>>((uint32_t*)Af, (int)(szX / 4));

  // 2 batches * 7 segment-slots * 4 heads/group * 32 query blocks = 1792
  dilated_attn_kernel<<<1792, 128, 0, stream>>>(Qf, Kf, Vf, Af);

  // LayerNorm -> f16 (reuse Q buffer)
  ln_kernel<<<M_, 256, 0, stream>>>(Af, ln_g, ln_b, Qf);

  // output projection -> f32 d_out
  gemm_wt_kernel<_Float16, float><<<ggrid, 256, 0, stream>>>(Qf, W16o, bo, (float*)d_out);
}
